// NonLocal_48309792145992
// MI455X (gfx1250) — compile-verified, matrix operand order checked by reading, not run on hardware
//
#include <hip/hip_runtime.h>

// ---------------------------------------------------------------------------
// Non-local attention block for MI455X (gfx1250, wave32, WMMA 16x16x32 f16)
// B=4, C=256, Ci=128, H=W=64 -> N=4096
// ---------------------------------------------------------------------------

typedef __attribute__((ext_vector_type(16))) _Float16 v16h;
typedef __attribute__((ext_vector_type(8)))  _Float16 v8h;
typedef __attribute__((ext_vector_type(8)))  float    v8f;
typedef __attribute__((ext_vector_type(4)))  int      v4i;

#define B_   4
#define C_   256
#define CI_  128
#define N_   4096
#define WMAT (CI_ * C_)                    // 32768 elements per weight matrix
#define SCALE_RSQRT 0.08838834764831845f   // 1/sqrt(128)

#if defined(__has_builtin)
#if __has_builtin(__builtin_amdgcn_global_load_async_to_lds_b128) && \
    __has_builtin(__builtin_amdgcn_s_wait_asynccnt)
#define USE_ASYNC_LDS 1
#endif
#endif
#ifndef USE_ASYNC_LDS
#define USE_ASYNC_LDS 0
#endif

__device__ __forceinline__ v16h cat8(v8h lo, v8h hi) {
  return __builtin_shufflevector(lo, hi, 0,1,2,3,4,5,6,7,8,9,10,11,12,13,14,15);
}

__device__ __forceinline__ v8f wmma16(v16h a, v16h b, v8f c) {
  // D = A(16x32 f16) * B(32x16 f16) + C(16x16 f32)
  return __builtin_amdgcn_wmma_f32_16x16x32_f16(
      false, a, false, b, (short)0, c, false, false);
}

// A/B operand fragment: two contiguous 16B runs at K = kb+8h and kb+16+8h.
__device__ __forceinline__ v16h frag(const _Float16* row, int kb, int h) {
  return cat8(*(const v8h*)(row + kb + 8 * h),
              *(const v8h*)(row + kb + 16 + 8 * h));
}

// 16-byte global -> LDS copy; async DMA on CDNA5 when available.
__device__ __forceinline__ void copy16_g2l(const _Float16* g, _Float16* l) {
#if USE_ASYNC_LDS
  __builtin_amdgcn_global_load_async_to_lds_b128(
      (__attribute__((address_space(1))) v4i*)g,
      (__attribute__((address_space(3))) v4i*)l, 0, 0);
#else
  *(v8h*)l = *(const v8h*)g;
#endif
}

__device__ __forceinline__ void wait_g2l() {
#if USE_ASYNC_LDS
  __builtin_amdgcn_s_wait_asynccnt(0);
#endif
}

// ---------------------------------------------------------------------------
// Stage 0a: convert all four weight matrices to f16 once (each 128x256-sized)
// dst layout: [w_theta | w_phi | w_g | w_out], each WMAT halfs, row-major.
// ---------------------------------------------------------------------------
__global__ __launch_bounds__(256) void wconv_kernel(
    const float* __restrict__ wt, const float* __restrict__ wp,
    const float* __restrict__ wg, const float* __restrict__ wo,
    _Float16* __restrict__ dst)
{
  const int i = blockIdx.x * 256 + threadIdx.x;   // 0 .. 32767
  dst[i]            = (_Float16)wt[i];
  dst[WMAT + i]     = (_Float16)wp[i];
  dst[2 * WMAT + i] = (_Float16)wg[i];
  dst[3 * WMAT + i] = (_Float16)wo[i];
}

// ---------------------------------------------------------------------------
// Stage 0b: transpose+convert x [B][C][N] f32 -> xT [B][N][C] f16 (LDS tiled)
// ---------------------------------------------------------------------------
__global__ __launch_bounds__(256) void transpose_kernel(
    const float* __restrict__ x, _Float16* __restrict__ xT)
{
  __shared__ _Float16 tile[32][34];          // padded: conflict-free columns
  const int id = blockIdx.x;
  const int nT = id & 127;
  const int cT = (id >> 7) & 7;
  const int b  = id >> 10;
  const int c0 = cT * 32, n0 = nT * 32;
  const int tn = threadIdx.x & 31;
  const int tr = threadIdx.x >> 5;           // 8 rows per pass
#pragma unroll
  for (int r = 0; r < 32; r += 8) {
    const int c = tr + r;
    tile[c][tn] = (_Float16)x[((size_t)b * C_ + c0 + c) * N_ + n0 + tn];
  }
  __syncthreads();
#pragma unroll
  for (int r = 0; r < 32; r += 8) {
    const int n = tr + r;
    xT[((size_t)b * N_ + n0 + n) * C_ + c0 + tn] = tile[tn][n];
  }
}

// ---------------------------------------------------------------------------
// Stage 1: 1x1-conv projections.  One wave computes one 16x16 tile of
//   P = W(Ci x C) @ x(C x N) + bias   (per batch, per projection)
// theta (pre-scaled by 1/sqrt(Ci)), phi stored [N][Ci]; g stored [Ci][N].
// All operands are f16 with contiguous 16B runs -> pure b128 loads.
// ---------------------------------------------------------------------------
__global__ __launch_bounds__(256) void proj_kernel(
    const _Float16* __restrict__ xT, const _Float16* __restrict__ wAll,
    const float* __restrict__ b_theta, const float* __restrict__ b_phi,
    const float* __restrict__ b_g,
    _Float16* __restrict__ thetaT, _Float16* __restrict__ phiT,
    _Float16* __restrict__ g_p)
{
  const int wave  = threadIdx.x >> 5;
  const int lane  = threadIdx.x & 31;
  const int id    = blockIdx.x * 8 + wave;
  const int nTile = id & 255;
  const int mTile = (id >> 8) & 7;
  const int b     = (id >> 11) & 3;
  const int p     = id >> 13;           // 0=theta 1=phi 2=g
  const int h     = lane >> 4;
  const int t     = lane & 15;
  const int nBase = nTile * 16;

  const float* bias = (p == 0) ? b_theta : (p == 1) ? b_phi : b_g;
  const _Float16* wRow = wAll + (size_t)p * WMAT + (size_t)(mTile * 16 + t) * C_;
  const _Float16* xRow = xT + ((size_t)b * N_ + nBase + t) * C_;

  v8f acc = {};
#pragma unroll
  for (int kb = 0; kb < C_; kb += 32)
    acc = wmma16(frag(wRow, kb, h), frag(xRow, kb, h), acc);

#pragma unroll
  for (int v = 0; v < 8; ++v) {
    const int m  = v + 8 * h;
    const int mG = mTile * 16 + m;
    float val = acc[v] + bias[mG];
    if (p == 0) val *= SCALE_RSQRT;     // fold softmax scale into theta
    const _Float16 hv = (_Float16)val;
    const size_t nG = (size_t)nBase + t;
    if (p == 0)      thetaT[((size_t)b * N_ + nG) * CI_ + mG] = hv;
    else if (p == 1) phiT  [((size_t)b * N_ + nG) * CI_ + mG] = hv;
    else             g_p   [((size_t)b * CI_ + mG) * N_ + nG] = hv;
  }
}

// ---------------------------------------------------------------------------
// Stage 2: flash attention.  Block = 4 waves sharing batch b; each wave owns
// 16 queries.  phi/g j-tiles double-buffered in LDS: async DMA for block
// jb+32 is issued before computing on jb, overlapping DMA with the 17 WMMAs
// + softmax of the current tile.  Row-sums via WMMA against a ones matrix.
// ---------------------------------------------------------------------------
__global__ __launch_bounds__(128) void attn_kernel(
    const _Float16* __restrict__ thetaT,
    const _Float16* __restrict__ phiT,
    const _Float16* __restrict__ g_p,
    _Float16* __restrict__ outT)
{
  __shared__ _Float16 phiTile[2][32 * CI_];   // [j][c]  2 x 8 KB
  __shared__ _Float16 gTile[2][CI_ * 32];     // [c][j]  2 x 8 KB
  __shared__ _Float16 pBuf[4 * 16 * 32];      // per-wave P transpose, 4 KB

  const int tid   = threadIdx.x;              // 0..127
  const int wave  = tid >> 5;                 // 0..3
  const int lane  = tid & 31;
  const int b     = blockIdx.x >> 6;          // 256 blocks = 4 b * 64 iBlk
  const int iBlk  = blockIdx.x & 63;
  const int iTile = iBlk * 4 + wave;          // 0..255
  const int h     = lane >> 4;
  const int t     = lane & 15;
  const int iBase = iTile * 16;
  _Float16* myP = pBuf + wave * (16 * 32);

  const _Float16* phiB = phiT + (size_t)b * N_ * CI_;
  const _Float16* gB   = g_p  + (size_t)b * CI_ * N_;

  // Cooperative stage of phi[jb..jb+31][*] (contiguous 8KB) and
  // g[*][jb..jb+31] (row c = tid, 32 halfs) into LDS buffer `buf`.
  auto stageTiles = [&](int buf, int jb) {
    const _Float16* ps = phiB + (size_t)jb * CI_ + tid * 32;
    _Float16*       pd = &phiTile[buf][tid * 32];
    copy16_g2l(ps, pd);           copy16_g2l(ps + 8, pd + 8);
    copy16_g2l(ps + 16, pd + 16); copy16_g2l(ps + 24, pd + 24);
    const _Float16* gs = gB + (size_t)tid * N_ + jb;
    _Float16*       gd = &gTile[buf][tid * 32];
    copy16_g2l(gs, gd);           copy16_g2l(gs + 8, gd + 8);
    copy16_g2l(gs + 16, gd + 16); copy16_g2l(gs + 24, gd + 24);
  };

  // Preload Q as 4 A-operands (K = Ci = 128); theta rows are contiguous.
  const _Float16* qRow = thetaT + ((size_t)b * N_ + iBase + t) * CI_;
  v16h qa[4];
#pragma unroll
  for (int ks = 0; ks < 4; ++ks) qa[ks] = frag(qRow, ks * 32, h);

  v16h ones;
#pragma unroll
  for (int e = 0; e < 16; ++e) ones[e] = (_Float16)1.0f;

  float mrow[8], lrow[8];
  v8f Oacc[8];
  const v8f vzero = {};
#pragma unroll
  for (int v = 0; v < 8; ++v) { mrow[v] = -1e30f; lrow[v] = 0.f; Oacc[v] = vzero; }

  stageTiles(0, 0);
  wait_g2l();
  __syncthreads();

  for (int jb = 0, it = 0; jb < N_; jb += 32, ++it) {
    const int cur = it & 1;
    if (jb + 32 < N_) stageTiles(cur ^ 1, jb + 32);   // overlap DMA w/ compute

    // ---- S = Q K^T for two 16-wide j tiles (8 WMMA), operands from LDS ----
    v8f s[2];
#pragma unroll
    for (int tt = 0; tt < 2; ++tt) {
      const _Float16* pRow = &phiTile[cur][(size_t)(16 * tt + t) * CI_];
      v8f sa = {};
#pragma unroll
      for (int ks = 0; ks < 4; ++ks)
        sa = wmma16(qa[ks], frag(pRow, ks * 32, h), sa);
      s[tt] = sa;
    }

    // ---- online softmax: row max via 16-lane xor reduction ----
    float alpha[8];
#pragma unroll
    for (int v = 0; v < 8; ++v) {
      float tm = fmaxf(s[0][v], s[1][v]);
#pragma unroll
      for (int off = 8; off >= 1; off >>= 1)
        tm = fmaxf(tm, __shfl_xor(tm, off, 32));
      const float mNew = fmaxf(mrow[v], tm);
      alpha[v] = __expf(mrow[v] - mNew);
      mrow[v]  = mNew;
      s[0][v] = __expf(s[0][v] - mNew);
      s[1][v] = __expf(s[1][v] - mNew);
    }
#pragma unroll
    for (int cb = 0; cb < 8; ++cb)
#pragma unroll
      for (int v = 0; v < 8; ++v) Oacc[cb][v] *= alpha[v];

    // ---- transpose P (D layout) -> A-operand layout via per-wave LDS ----
#pragma unroll
    for (int tt = 0; tt < 2; ++tt)
#pragma unroll
      for (int v = 0; v < 8; ++v) {
        const int m = v + 8 * h;
        myP[m * 32 + 16 * tt + t] = (_Float16)s[tt][v];
      }
    asm volatile("s_wait_dscnt 0" ::: "memory");   // wave-local LDS RAW fence
    const v16h pa = frag(myP + t * 32, 0, h);

    // ---- row-sum of P with one WMMA against ones (replaces shuffles) ----
    const v8f rsum = wmma16(pa, ones, vzero);
#pragma unroll
    for (int v = 0; v < 8; ++v) lrow[v] = lrow[v] * alpha[v] + rsum[v];

    // ---- O += P @ g^T (8 WMMA across Ci), B operands from LDS ----
#pragma unroll
    for (int cb = 0; cb < 8; ++cb) {
      const _Float16* gRow = &gTile[cur][(size_t)(cb * 16 + t) * 32];
      Oacc[cb] = wmma16(pa, frag(gRow, 0, h), Oacc[cb]);
    }

    wait_g2l();        // next-tile DMA done (overlapped with compute above)
    __syncthreads();   // all waves done with `cur` before it is restaged
  }

  // ---- finalize: divide by row-sum, store O as [N][Ci] f16 ----
  float linv[8];
#pragma unroll
  for (int v = 0; v < 8; ++v) linv[v] = 1.0f / lrow[v];
#pragma unroll
  for (int cb = 0; cb < 8; ++cb)
#pragma unroll
    for (int v = 0; v < 8; ++v) {
      const int m = v + 8 * h;
      outT[((size_t)b * N_ + iBase + m) * CI_ + cb * 16 + t] =
          (_Float16)(Oacc[cb][v] * linv[v]);
    }
}

// ---------------------------------------------------------------------------
// Stage 3: y = w_out @ O + b_out + x  (16x16 tiles, K=Ci=128 -> 4 WMMA)
// ---------------------------------------------------------------------------
__global__ __launch_bounds__(256) void out_kernel(
    const _Float16* __restrict__ w16, const float* __restrict__ b_out,
    const _Float16* __restrict__ outT,
    const float* __restrict__ x, float* __restrict__ y)
{
  const int wave  = threadIdx.x >> 5;
  const int lane  = threadIdx.x & 31;
  const int id    = blockIdx.x * 8 + wave;
  const int nTile = id & 255;
  const int oTile = (id >> 8) & 15;
  const int b     = id >> 12;
  const int h     = lane >> 4;
  const int t     = lane & 15;
  const int nBase = nTile * 16;

  const _Float16* wRow = w16 + (size_t)(oTile * 16 + t) * CI_;
  const _Float16* oRow = outT + ((size_t)b * N_ + nBase + t) * CI_;

  v8f acc = {};
#pragma unroll
  for (int ks = 0; ks < 4; ++ks)
    acc = wmma16(frag(wRow, ks * 32, h), frag(oRow, ks * 32, h), acc);

  const size_t xbase = (size_t)b * C_ * N_;
#pragma unroll
  for (int v = 0; v < 8; ++v) {
    const int m  = v + 8 * h;
    const int oG = oTile * 16 + m;
    const size_t idx = xbase + (size_t)oG * N_ + nBase + t;
    y[idx] = acc[v] + b_out[oG] + x[idx];      // fused bias + residual
  }
}

// ---------------------------------------------------------------------------
extern "C" void kernel_launch(void* const* d_in, const int* in_sizes, int n_in,
                              void* d_out, int out_size, void* d_ws, size_t ws_size,
                              hipStream_t stream) {
  (void)in_sizes; (void)n_in; (void)out_size; (void)ws_size;
  const float* x       = (const float*)d_in[0];
  const float* w_g     = (const float*)d_in[1];
  const float* b_g     = (const float*)d_in[2];
  const float* w_theta = (const float*)d_in[3];
  const float* b_theta = (const float*)d_in[4];
  const float* w_phi   = (const float*)d_in[5];
  const float* b_phi   = (const float*)d_in[6];
  const float* w_out   = (const float*)d_in[7];
  const float* b_out   = (const float*)d_in[8];
  float* y = (float*)d_out;

  _Float16* ws = (_Float16*)d_ws;
  const size_t per = (size_t)B_ * N_ * CI_;   // 2 Mi halfs
  _Float16* thetaT = ws;                      // [B][N][Ci]
  _Float16* phiT   = ws + per;                // [B][N][Ci]
  _Float16* g_p    = ws + 2 * per;            // [B][Ci][N]
  _Float16* xT     = ws + 3 * per;            // [B][N][C] (4 Mi halfs)
  _Float16* outT   = xT;                      // aliases xT (dead after proj)
  _Float16* wAll   = ws + 5 * per;            // 4 converted weight matrices

  // 32768 weight elements / 256 threads = 128 blocks
  wconv_kernel<<<128, 256, 0, stream>>>(w_theta, w_phi, w_g, w_out, wAll);
  // 4 batch * 8 cTiles * 128 nTiles = 4096 blocks
  transpose_kernel<<<4096, 256, 0, stream>>>(x, xT);
  // 3 proj * 4 batch * 8 mTiles * 256 nTiles = 24576 wave-tiles / 8 waves
  proj_kernel<<<3072, 256, 0, stream>>>(xT, wAll, b_theta, b_phi, b_g,
                                        thetaT, phiT, g_p);
  // 4 batch * 64 iBlocks = 256 blocks (4 i-tiles each, 128 threads)
  attn_kernel<<<256, 128, 0, stream>>>(thetaT, phiT, g_p, outT);
  // 4 batch * 16 oTiles * 256 nTiles = 16384 wave-tiles / 8 waves
  out_kernel<<<2048, 256, 0, stream>>>(wAll + 3 * (size_t)WMAT, b_out, outT, x, y);
}
